// E3TransformerBlock_61254823575809
// MI455X (gfx1250) — matrix operand order repaired and checked by reading, not compiled, not run on hardware
//
#include <hip/hip_runtime.h>

// ---------------------------------------------------------------------------
// E3-equivariant transformer block for MI455X (gfx1250, wave32, WMMA bf16).
// Dense GEMMs on v_wmma_f32_16x16x32_bf16; per-wave 16-row tiles chained
// through LDS (s_wait_dscnt); hs tiles staged into LDS via the Tensor Data
// Mover (tensor_load_to_lds + s_wait_tensorcnt); weight prefetch into cache.
// ---------------------------------------------------------------------------

#define N_NODES 20000
#define N_EDGES 400000

typedef __bf16 bf16;
typedef __attribute__((ext_vector_type(16))) __bf16 bf16x16;
typedef __attribute__((ext_vector_type(8)))  __bf16 bf16x8;
typedef __attribute__((ext_vector_type(8)))  float  f32x8;
typedef unsigned int u32x4 __attribute__((ext_vector_type(4)));
typedef int i32x4 __attribute__((ext_vector_type(4)));
typedef int i32x8 __attribute__((ext_vector_type(8)));

// ---------------- workspace layout (bytes) ----------------
constexpr size_t OFF_WT_E1  = 0;                                  // 128 x 288 bf16 (K=264 pad 288)
constexpr size_t OFF_WT_E2  = OFF_WT_E1  + (size_t)128*288*2;
constexpr size_t OFF_WT_N1  = OFF_WT_E2  + (size_t)128*128*2;
constexpr size_t OFF_WT_N2  = OFF_WT_N1  + (size_t)128*128*2;
constexpr size_t OFF_WT_TP1 = OFF_WT_N2  + (size_t)128*128*2;
constexpr size_t OFF_WT_TP2 = OFF_WT_TP1 + (size_t)128*128*2;
constexpr size_t OFF_WT_Q   = OFF_WT_TP2 + (size_t)128*128*2;
constexpr size_t OFF_WT_K   = OFF_WT_Q   + (size_t)128*128*2;
constexpr size_t OFF_WT_VS  = OFF_WT_K   + (size_t)128*128*2;
constexpr size_t OFF_WT_VV  = OFF_WT_VS  + (size_t)128*128*2;     // 64 x 128 bf16
constexpr size_t OFF_WT_F1  = OFF_WT_VV  + (size_t)64*128*2;
constexpr size_t OFF_WT_F2  = OFF_WT_F1  + (size_t)128*128*2;
constexpr size_t OFF_WT_F3  = OFF_WT_F2  + (size_t)128*128*2;
constexpr size_t OFF_H      = OFF_WT_F3  + (size_t)128*128*2;     // N x 128 f32
constexpr size_t OFF_Q      = OFF_H      + (size_t)N_NODES*128*4; // N x 128 f32
constexpr size_t OFF_HS     = OFF_Q      + (size_t)N_NODES*128*4; // E x 128 bf16
constexpr size_t OFF_LOGITS = OFF_HS     + (size_t)N_EDGES*128*2; // E x 4 f32
constexpr size_t OFF_WEXP   = OFF_LOGITS + (size_t)N_EDGES*4*4;   // E x 4 f32
constexpr size_t OFF_SEGMAX = OFF_WEXP   + (size_t)N_EDGES*4*4;   // N x 4 int (ordered keys)
constexpr size_t OFF_DENOM  = OFF_SEGMAX + (size_t)N_NODES*4*4;   // N x 4 f32
constexpr size_t OFF_OUTS   = OFF_DENOM  + (size_t)N_NODES*4*4;   // N x 128 f32
constexpr size_t OFF_OUTV   = OFF_OUTS   + (size_t)N_NODES*128*4; // N x 24 f32

// ---------------- small device helpers ----------------
__device__ __forceinline__ void lds_fence() {
  asm volatile("s_wait_dscnt 0" ::: "memory");
}

__device__ __forceinline__ float silu_(float x) { return x / (1.0f + __expf(-x)); }

__device__ __forceinline__ int f2key(float f) {
  int i = __float_as_int(f);
  return (i >= 0) ? i : (i ^ 0x7fffffff);
}
__device__ __forceinline__ float key2f(int k) {
  return __int_as_float((k >= 0) ? k : (k ^ 0x7fffffff));
}

// ---- Tensor Data Mover: DMA a contiguous 4 KB tile (global -> LDS). ----
// D# per ISA 8.3/8.4: 1-row tile, data_size=8B, tile_dim0=512 units.
// LDS byte address = low 32 bits of the flat shared-aperture address.
__device__ __forceinline__ void tdm_load_4kb(const void* gsrc, void* lds_dst) {
  unsigned lds_addr = (unsigned)(size_t)lds_dst;
  unsigned long long ga = (unsigned long long)(size_t)gsrc;
  const unsigned dim0 = 0x40000000u;            // tensor_dim0 (8B units), large
  u32x4 g0;
  g0[0] = 1u;                                   // count=1, user descriptor
  g0[1] = lds_addr;                             // lds_addr[31:0]
  g0[2] = (unsigned)(ga & 0xffffffffull);       // global_addr[31:0]
  g0[3] = (unsigned)((ga >> 32) & 0x01ffffffull) | (2u << 30); // addr[56:32] | type=2
  i32x8 g1;
  g1[0] = (int)(3u << 16);                      // wg_mask=0, data_size=3 (8 bytes)
  g1[1] = (int)((dim0 & 0xffffu) << 16);        // barrier_addr=0 | tensor_dim0[15:0]
  g1[2] = (int)((dim0 >> 16) | (1u << 16));     // tensor_dim0[31:16] | tensor_dim1=1 lo
  g1[3] = (int)(512u << 16);                    // tensor_dim1 hi=0 | tile_dim0=512
  g1[4] = 0;                                    // tile_dim1=0, tile_dim2=0 (unused)
  g1[5] = (int)dim0;                            // tensor_dim0_stride[31:0]
  g1[6] = 0;                                    // stride hi | dim1_stride lo
  g1[7] = 0;
  i32x4 gz4 = {0, 0, 0, 0};
  i32x8 gz8 = {0, 0, 0, 0, 0, 0, 0, 0};
  __builtin_amdgcn_tensor_load_to_lds(g0, g1, gz4, gz4, gz8, 0);
}
__device__ __forceinline__ void tensor_wait() {
  __builtin_amdgcn_s_wait_tensorcnt(0);
  asm volatile("" ::: "memory");
}

// A-fragment: 16x32 bf16 tile, row-major (stride lda elements).
// ISA 7.12.2: lanes 0-15 (row m=lane) hold K = k0+0..7 and k0+16..23;
//             lanes 16-31 (row m=lane-16) hold K = k0+8..15 and k0+24..31.
__device__ __forceinline__ bf16x16 load_fragA(const bf16* base, int lda, int k0, int lane) {
  int m  = lane & 15;
  int kb = k0 + ((lane & 16) ? 8 : 0);
  const bf16* p = base + m * lda + kb;
  bf16x8 lo = *(const bf16x8*)(p);
  bf16x8 hi = *(const bf16x8*)(p + 16);
  bf16x16 r;
#pragma unroll
  for (int e = 0; e < 8; ++e) { r[e] = lo[e]; r[8 + e] = hi[e]; }
  return r;
}

// B-fragment from transposed weights WT (row n holds W[:,n], stride ldb = Kpad):
// lane n<16 -> column n, K = k0+0..7,16..23; lane n+16 -> column n, K = k0+8..15,24..31.
__device__ __forceinline__ bf16x16 load_fragB(const bf16* wt, int ldb, int k0, int lane) {
  int n  = lane & 15;
  int kb = k0 + ((lane & 16) ? 8 : 0);
  const bf16* p = wt + n * ldb + kb;
  bf16x8 lo = *(const bf16x8*)(p);
  bf16x8 hi = *(const bf16x8*)(p + 16);
  bf16x16 r;
#pragma unroll
  for (int e = 0; e < 8; ++e) { r[e] = lo[e]; r[8 + e] = hi[e]; }
  return r;
}

template <int NT>
__device__ __forceinline__ void gemm_tiles(const bf16* A, int lda, const bf16* WT, int ldb,
                                           int K, f32x8* acc, int lane) {
  for (int k0 = 0; k0 < K; k0 += 32) {
    bf16x16 a = load_fragA(A, lda, k0, lane);
#pragma unroll
    for (int t = 0; t < NT; ++t) {
      bf16x16 b = load_fragB(WT + (size_t)(t * 16) * ldb, ldb, k0, lane);
      acc[t] = __builtin_amdgcn_wmma_f32_16x16x32_bf16(false, a, false, b, (short)0,
                                                       acc[t], false, false);
    }
  }
}

template <int NT>
__device__ __forceinline__ void zero_acc(f32x8* a) {
#pragma unroll
  for (int t = 0; t < NT; ++t)
#pragma unroll
    for (int v = 0; v < 8; ++v) a[t][v] = 0.0f;
}

// D-tile store to LDS (bf16), optional silu. ISA C/D layout:
// lane<16: (M=v, N=lane); lane>=16: (M=8+v, N=lane-16).
__device__ __forceinline__ void store_tile_lds(bf16* dst, int ld, int n0, const f32x8& c,
                                               int lane, bool act) {
  int n  = n0 + (lane & 15);
  int mb = (lane & 16) ? 8 : 0;
#pragma unroll
  for (int v = 0; v < 8; ++v) {
    float x = c[v];
    if (act) x = silu_(x);
    dst[(mb + v) * ld + n] = (bf16)x;
  }
}

// Warm WGP$/L2 with a weight matrix (emits global_prefetch_b8).
__device__ __forceinline__ void prefetch_weights(const bf16* wt, int bytes, int tid) {
  int off = tid * 128;
  if (off < bytes) __builtin_prefetch((const char*)wt + off, 0, 0);
}

// ---------------- kernel 0: weight transpose/convert + scratch init ----------------
__global__ void prep_kernel(const float* We1, const float* We2, const float* Wn1,
                            const float* Wn2, const float* Wtp1, const float* Wtp2,
                            const float* Wq, const float* Wk, const float* Wvs,
                            const float* Wvv, const float* Wf1, const float* Wf2,
                            const float* Wf3, char* ws) {
  int tid = blockIdx.x * blockDim.x + threadIdx.x;
  int nth = gridDim.x * blockDim.x;
  {
    bf16* wt = (bf16*)(ws + OFF_WT_E1);
    for (int i = tid; i < 128 * 288; i += nth) {
      int n = i / 288, k = i - n * 288;
      wt[i] = (bf16)((k < 264) ? We1[k * 128 + n] : 0.0f);
    }
  }
  const float* src[11] = {We2, Wn1, Wn2, Wtp1, Wtp2, Wq, Wk, Wvs, Wf1, Wf2, Wf3};
  const size_t dst[11] = {OFF_WT_E2, OFF_WT_N1, OFF_WT_N2, OFF_WT_TP1, OFF_WT_TP2,
                          OFF_WT_Q, OFF_WT_K, OFF_WT_VS, OFF_WT_F1, OFF_WT_F2, OFF_WT_F3};
  for (int m = 0; m < 11; ++m) {
    bf16* wt = (bf16*)(ws + dst[m]);
    const float* W = src[m];
    for (int i = tid; i < 128 * 128; i += nth) {
      int n = i >> 7, k = i & 127;
      wt[i] = (bf16)W[k * 128 + n];
    }
  }
  {
    bf16* wt = (bf16*)(ws + OFF_WT_VV);
    for (int i = tid; i < 64 * 128; i += nth) {
      int n = i >> 7, k = i & 127;
      wt[i] = (bf16)Wvv[k * 64 + n];
    }
  }
  int*   segmax = (int*)(ws + OFF_SEGMAX);
  float* denom  = (float*)(ws + OFF_DENOM);
  for (int i = tid; i < N_NODES * 4; i += nth) { segmax[i] = (int)0x80000000; denom[i] = 0.f; }
  float* outs = (float*)(ws + OFF_OUTS);
  for (int i = tid; i < N_NODES * 128; i += nth) outs[i] = 0.f;
  float* outv = (float*)(ws + OFF_OUTV);
  for (int i = tid; i < N_NODES * 24; i += nth) outv[i] = 0.f;
}

// ---------------- kernel 1: node MLP -> h, q ----------------
__global__ void __launch_bounds__(128) node_kernel(const float* nf, const float* b1,
                                                   const float* b2, char* ws) {
  __shared__ __align__(16) bf16 sIn[4][16 * 128];
  __shared__ __align__(16) bf16 sT[4][16 * 128];
  const bf16* wt_n1 = (const bf16*)(ws + OFF_WT_N1);
  const bf16* wt_n2 = (const bf16*)(ws + OFF_WT_N2);
  const bf16* wt_q  = (const bf16*)(ws + OFF_WT_Q);
  float* h = (float*)(ws + OFF_H);
  float* q = (float*)(ws + OFF_Q);
  int lane = threadIdx.x & 31, wv = threadIdx.x >> 5;
  int tile = blockIdx.x * 4 + wv;
  bool valid = tile < (N_NODES / 16);
  if (!valid) tile = 0;
  int row0 = tile * 16;
  bf16* In = sIn[wv];
  bf16* T  = sT[wv];
  for (int i = lane; i < 16 * 128; i += 32) {
    int r = i >> 7, c = i & 127;
    In[i] = (bf16)nf[(size_t)(row0 + r) * 128 + c];
  }
  lds_fence();
  int nn = lane & 15, mb = (lane & 16) ? 8 : 0;
  f32x8 acc[8];
#pragma unroll
  for (int t = 0; t < 8; ++t) {
    float b = b1[t * 16 + nn];
#pragma unroll
    for (int v = 0; v < 8; ++v) acc[t][v] = b;
  }
  gemm_tiles<8>(In, 128, wt_n1, 128, 128, acc, lane);
#pragma unroll
  for (int t = 0; t < 8; ++t) store_tile_lds(T, 128, t * 16, acc[t], lane, true);
  lds_fence();
#pragma unroll
  for (int t = 0; t < 8; ++t) {
    float b = b2[t * 16 + nn];
#pragma unroll
    for (int v = 0; v < 8; ++v) acc[t][v] = b;
  }
  gemm_tiles<8>(T, 128, wt_n2, 128, 128, acc, lane);
#pragma unroll
  for (int t = 0; t < 8; ++t) {
    int n = t * 16 + nn;
#pragma unroll
    for (int v = 0; v < 8; ++v) {
      float x = acc[t][v];
      if (valid) h[(size_t)(row0 + mb + v) * 128 + n] = x;
      In[(mb + v) * 128 + n] = (bf16)x;
    }
  }
  lds_fence();
  zero_acc<8>(acc);
  gemm_tiles<8>(In, 128, wt_q, 128, 128, acc, lane);
  if (valid) {
#pragma unroll
    for (int t = 0; t < 8; ++t) {
      int n = t * 16 + nn;
#pragma unroll
      for (int v = 0; v < 8; ++v) q[(size_t)(row0 + mb + v) * 128 + n] = acc[t][v];
    }
  }
}

// ---------------- kernel 2: edge chain -> hs (bf16) ----------------
__global__ void __launch_bounds__(64) edge_hs_kernel(const float* pos, const float* nf,
                                                     const int* snd, const int* rcv,
                                                     char* ws) {
  __shared__ __align__(16) bf16 sE[2][16 * 288];
  __shared__ __align__(16) bf16 sA[2][16 * 128];
  __shared__ __align__(16) bf16 sB[2][16 * 128];
  const bf16* wt_e1  = (const bf16*)(ws + OFF_WT_E1);
  const bf16* wt_e2  = (const bf16*)(ws + OFF_WT_E2);
  const bf16* wt_tp1 = (const bf16*)(ws + OFF_WT_TP1);
  const bf16* wt_tp2 = (const bf16*)(ws + OFF_WT_TP2);
  const float* h = (const float*)(ws + OFF_H);
  bf16* hs = (bf16*)(ws + OFF_HS);
  int lane = threadIdx.x & 31, wv = threadIdx.x >> 5;
  int e0 = (blockIdx.x * 2 + wv) * 16;
  bf16* E = sE[wv];
  bf16* A = sA[wv];
  bf16* B = sB[wv];
  // warm the weight working set (global_prefetch_b8)
  prefetch_weights(wt_e1, 128 * 288 * 2, threadIdx.x);
  prefetch_weights(wt_tp2, 128 * 128 * 2, threadIdx.x);
  // build e_in = [ |r|^2 (8) | nf[send] (128) | nf[recv] (128) | zero pad (24) ]
  for (int i = lane; i < 16 * 288; i += 32) {
    int r = i / 288, c = i - r * 288;
    int e = e0 + r;
    float val;
    if (c < 8) {
      int s = snd[e], rr = rcv[e];
      const float* ps = pos + (size_t)(s * 8 + c) * 3;
      const float* pr = pos + (size_t)(rr * 8 + c) * 3;
      float dx = pr[0] - ps[0], dy = pr[1] - ps[1], dz = pr[2] - ps[2];
      val = dx * dx + dy * dy + dz * dz;
    } else if (c < 136) {
      val = nf[(size_t)snd[e] * 128 + (c - 8)];
    } else if (c < 264) {
      val = nf[(size_t)rcv[e] * 128 + (c - 136)];
    } else {
      val = 0.f;
    }
    E[i] = (bf16)val;
  }
  lds_fence();
  f32x8 acc[8];
  zero_acc<8>(acc);
  gemm_tiles<8>(E, 288, wt_e1, 288, 288, acc, lane);       // t = silu(e_in @ We1)
#pragma unroll
  for (int t = 0; t < 8; ++t) store_tile_lds(A, 128, t * 16, acc[t], lane, true);
  lds_fence();
  zero_acc<8>(acc);
  gemm_tiles<8>(A, 128, wt_e2, 128, 128, acc, lane);       // s_e = t @ We2
#pragma unroll
  for (int t = 0; t < 8; ++t) store_tile_lds(B, 128, t * 16, acc[t], lane, false);
  lds_fence();
  zero_acc<8>(acc);
  gemm_tiles<8>(B, 128, wt_tp1, 128, 128, acc, lane);      // a = silu(s_e @ Wtp1)
#pragma unroll
  for (int t = 0; t < 8; ++t) store_tile_lds(A, 128, t * 16, acc[t], lane, true);
  lds_fence();
  zero_acc<8>(acc);
  gemm_tiles<8>(A, 128, wt_tp2, 128, 128, acc, lane);      // w_e = silu(a @ Wtp2)
  int nn = lane & 15, mb = (lane & 16) ? 8 : 0;
#pragma unroll
  for (int t = 0; t < 8; ++t) {
    int n = t * 16 + nn;
#pragma unroll
    for (int v = 0; v < 8; ++v) {
      int e = e0 + mb + v;
      float x = silu_(acc[t][v]) * h[(size_t)snd[e] * 128 + n];   // hs = h[send]*w_e
      hs[(size_t)e * 128 + n] = (bf16)x;
    }
  }
}

// ---------------- kernel 3: k = hs@Wk, logits, segment max ----------------
__global__ void __launch_bounds__(128) edge_logits_kernel(const int* rcv, char* ws) {
  __shared__ __align__(16) bf16  sH[4][16 * 128];
  __shared__ __align__(16) float sK[4][16 * 128];
  const bf16* hs   = (const bf16*)(ws + OFF_HS);
  const bf16* wt_k = (const bf16*)(ws + OFF_WT_K);
  const float* q   = (const float*)(ws + OFF_Q);
  float* logits = (float*)(ws + OFF_LOGITS);
  int* segmax   = (int*)(ws + OFF_SEGMAX);
  int lane = threadIdx.x & 31, wv = threadIdx.x >> 5;
  int e0 = (blockIdx.x * 4 + wv) * 16;
  bf16* H = sH[wv];
  float* Kt = sK[wv];
  // async DMA of the contiguous 4KB hs tile into LDS via the TDM
  tdm_load_4kb(hs + (size_t)e0 * 128, H);
  prefetch_weights(wt_k, 128 * 128 * 2, threadIdx.x);
  tensor_wait();
  f32x8 acc[8];
  zero_acc<8>(acc);
  gemm_tiles<8>(H, 128, wt_k, 128, 128, acc, lane);
  int nn = lane & 15, mb = (lane & 16) ? 8 : 0;
#pragma unroll
  for (int t = 0; t < 8; ++t) {
    int n = t * 16 + nn;
#pragma unroll
    for (int v = 0; v < 8; ++v) Kt[(mb + v) * 128 + n] = acc[t][v];
  }
  lds_fence();
  for (int p = lane; p < 64; p += 32) {   // 16 edges x 4 heads
    int m = p >> 2, hh = p & 3;
    int e = e0 + m;
    int r = rcv[e];
    const float* qp = q + (size_t)r * 128 + hh * 32;
    const float* kp = Kt + m * 128 + hh * 32;
    float d = 0.f;
#pragma unroll
    for (int j = 0; j < 32; ++j) d += qp[j] * kp[j];
    d *= 0.1767766952966369f;   // 1/sqrt(32)
    logits[(size_t)e * 4 + hh] = d;
    atomicMax(&segmax[r * 4 + hh], f2key(d));
  }
}

// ---------------- kernel 4: wexp + denominator ----------------
__global__ void wexp_kernel(const int* rcv, char* ws) {
  const float* logits = (const float*)(ws + OFF_LOGITS);
  const int* segmax   = (const int*)(ws + OFF_SEGMAX);
  float* wexp  = (float*)(ws + OFF_WEXP);
  float* denom = (float*)(ws + OFF_DENOM);
  int idx0 = blockIdx.x * blockDim.x + threadIdx.x;
  int stride = gridDim.x * blockDim.x;
  for (int idx = idx0; idx < N_EDGES * 4; idx += stride) {
    int e = idx >> 2, hh = idx & 3;
    int r = rcv[e];
    float mx = key2f(segmax[r * 4 + hh]);
    float w = __expf(logits[idx] - mx);
    wexp[idx] = w;
    atomicAdd(&denom[r * 4 + hh], w);
  }
}

// ---------------- kernel 5: values + attention-weighted scatter ----------------
__global__ void __launch_bounds__(128) edge_scatter_kernel(const int* snd, const int* rcv,
                                                           const float* pos, char* ws) {
  __shared__ __align__(16) bf16  sH[4][16 * 128];
  __shared__ __align__(16) float sMix[4][16 * 64];
  __shared__ float sAttn[4][64];
  const bf16* hs    = (const bf16*)(ws + OFF_HS);
  const bf16* wt_vs = (const bf16*)(ws + OFF_WT_VS);
  const bf16* wt_vv = (const bf16*)(ws + OFF_WT_VV);
  const float* wexp  = (const float*)(ws + OFF_WEXP);
  const float* denom = (const float*)(ws + OFF_DENOM);
  float* outs = (float*)(ws + OFF_OUTS);
  float* outv = (float*)(ws + OFF_OUTV);
  int lane = threadIdx.x & 31, wv = threadIdx.x >> 5;
  int e0 = (blockIdx.x * 4 + wv) * 16;
  bf16* H = sH[wv];
  // kick off the hs tile DMA, overlap with attention-weight computation
  tdm_load_4kb(hs + (size_t)e0 * 128, H);
  for (int p = lane; p < 64; p += 32) {
    int m = p >> 2, hh = p & 3;
    int e = e0 + m;
    int r = rcv[e];
    sAttn[wv][p] = wexp[(size_t)e * 4 + hh] / (denom[r * 4 + hh] + 1e-9f);
  }
  lds_fence();
  tensor_wait();
  int nn = lane & 15, mb = (lane & 16) ? 8 : 0;
  // v_s = hs @ Wvs, scatter attn*v_s
  f32x8 acc[8];
  zero_acc<8>(acc);
  gemm_tiles<8>(H, 128, wt_vs, 128, 128, acc, lane);
#pragma unroll
  for (int t = 0; t < 8; ++t) {
    int n = t * 16 + nn;
    int hh = n >> 5;
#pragma unroll
    for (int v = 0; v < 8; ++v) {
      int m = mb + v;
      int e = e0 + m;
      float val = acc[t][v] * sAttn[wv][m * 4 + hh];
      atomicAdd(&outs[(size_t)rcv[e] * 128 + n], val);
    }
  }
  // mix = hs @ Wvv (E,64) -> LDS
  f32x8 acc4[4];
  zero_acc<4>(acc4);
  gemm_tiles<4>(H, 128, wt_vv, 128, 128, acc4, lane);
#pragma unroll
  for (int t = 0; t < 4; ++t) {
    int n = t * 16 + nn;
#pragma unroll
    for (int v = 0; v < 8; ++v) sMix[wv][(mb + v) * 64 + n] = acc4[t][v];
  }
  lds_fence();
  // v_v[h,o,d] = sum_c mix[h,o,c]*edge_vec[c,d]; scatter attn*v_v
  for (int p = lane; p < 16 * 24; p += 32) {
    int m = p / 24, j = p - m * 24;      // j = (h*2+o)*3 + d
    int ch = j / 3, d = j - ch * 3;
    int hh = ch >> 1, o = ch & 1;
    int e = e0 + m;
    int s = snd[e], r = rcv[e];
    const float* mrow = sMix[wv] + m * 64 + hh * 16 + o * 8;
    float a = 0.f;
#pragma unroll
    for (int c = 0; c < 8; ++c) {
      float ev = pos[(size_t)(r * 8 + c) * 3 + d] - pos[(size_t)(s * 8 + c) * 3 + d];
      a += mrow[c] * ev;
    }
    atomicAdd(&outv[(size_t)r * 24 + j], a * sAttn[wv][m * 4 + hh]);
  }
}

// ---------------- kernel 6: final MLP + residuals ----------------
__global__ void __launch_bounds__(128) final_kernel(const float* nf, const float* pos,
                                                    char* ws, float* out) {
  __shared__ __align__(16) bf16 sIn[4][16 * 128];
  __shared__ __align__(16) bf16 sT[4][16 * 128];
  const float* outs = (const float*)(ws + OFF_OUTS);
  const float* outv = (const float*)(ws + OFF_OUTV);
  const bf16* wt_f1 = (const bf16*)(ws + OFF_WT_F1);
  const bf16* wt_f2 = (const bf16*)(ws + OFF_WT_F2);
  const bf16* wt_f3 = (const bf16*)(ws + OFF_WT_F3);
  float* vec_out  = out;                           // (N, 8, 3)
  float* feat_out = out + (size_t)N_NODES * 24;    // (N, 128)
  int lane = threadIdx.x & 31, wv = threadIdx.x >> 5;
  int tile = blockIdx.x * 4 + wv;
  bool valid = tile < (N_NODES / 16);
  if (!valid) tile = 0;
  int row0 = tile * 16;
  bf16* In = sIn[wv];
  bf16* T  = sT[wv];
  for (int i = lane; i < 16 * 128; i += 32) {
    int r = i >> 7, c = i & 127;
    In[i] = (bf16)outs[(size_t)(row0 + r) * 128 + c];
  }
  lds_fence();
  f32x8 acc[8];
  zero_acc<8>(acc);
  gemm_tiles<8>(In, 128, wt_f1, 128, 128, acc, lane);
#pragma unroll
  for (int t = 0; t < 8; ++t) store_tile_lds(T, 128, t * 16, acc[t], lane, true);
  lds_fence();
  zero_acc<8>(acc);
  gemm_tiles<8>(T, 128, wt_f2, 128, 128, acc, lane);
#pragma unroll
  for (int t = 0; t < 8; ++t) store_tile_lds(In, 128, t * 16, acc[t], lane, true);
  lds_fence();
  zero_acc<8>(acc);
  gemm_tiles<8>(In, 128, wt_f3, 128, 128, acc, lane);
  int nn = lane & 15, mb = (lane & 16) ? 8 : 0;
  if (valid) {
#pragma unroll
    for (int t = 0; t < 8; ++t) {
      int n = t * 16 + nn;
#pragma unroll
      for (int v = 0; v < 8; ++v) {
        size_t idx = (size_t)(row0 + mb + v) * 128 + n;
        feat_out[idx] = acc[t][v] + nf[idx];
      }
    }
    for (int i = lane; i < 16 * 24; i += 32) {
      size_t idx = (size_t)row0 * 24 + i;
      vec_out[idx] = pos[idx] + outv[idx];
    }
  }
}

// ---------------- host entry ----------------
extern "C" void kernel_launch(void* const* d_in, const int* in_sizes, int n_in,
                              void* d_out, int out_size, void* d_ws, size_t ws_size,
                              hipStream_t stream) {
  (void)in_sizes; (void)n_in; (void)out_size; (void)ws_size;
  const float* pos  = (const float*)d_in[0];
  const float* nf   = (const float*)d_in[1];
  const int*   snd  = (const int*)d_in[2];
  const int*   rcv  = (const int*)d_in[3];
  const float* We1  = (const float*)d_in[4];
  const float* We2  = (const float*)d_in[5];
  const float* Wn1  = (const float*)d_in[6];
  const float* b1   = (const float*)d_in[7];
  const float* Wn2  = (const float*)d_in[8];
  const float* b2   = (const float*)d_in[9];
  const float* Wtp1 = (const float*)d_in[10];
  const float* Wtp2 = (const float*)d_in[11];
  const float* Wq   = (const float*)d_in[12];
  const float* Wk   = (const float*)d_in[13];
  const float* Wvs  = (const float*)d_in[14];
  const float* Wvv  = (const float*)d_in[15];
  const float* Wf1  = (const float*)d_in[16];
  const float* Wf2  = (const float*)d_in[17];
  const float* Wf3  = (const float*)d_in[18];
  char* ws  = (char*)d_ws;
  float* out = (float*)d_out;

  prep_kernel<<<512, 256, 0, stream>>>(We1, We2, Wn1, Wn2, Wtp1, Wtp2, Wq, Wk, Wvs, Wvv,
                                       Wf1, Wf2, Wf3, ws);
  node_kernel<<<(N_NODES / 16 + 3) / 4, 128, 0, stream>>>(nf, b1, b2, ws);
  edge_hs_kernel<<<N_EDGES / 32, 64, 0, stream>>>(pos, nf, snd, rcv, ws);
  edge_logits_kernel<<<N_EDGES / 64, 128, 0, stream>>>(rcv, ws);
  wexp_kernel<<<2048, 256, 0, stream>>>(rcv, ws);
  edge_scatter_kernel<<<N_EDGES / 64, 128, 0, stream>>>(snd, rcv, pos, ws);
  final_kernel<<<(N_NODES / 16 + 3) / 4, 128, 0, stream>>>(nf, pos, ws, out);
}
